// Generator_43576738185307
// MI455X (gfx1250) — compile-verified, matrix operand order checked by reading, not run on hardware
//
#include <hip/hip_runtime.h>
#include <hip/hip_bf16.h>
#include <math.h>

// ---------------------------------------------------------------------------
// Types for CDNA5 WMMA (wave32, 16x16x32 bf16 -> f32)
// ---------------------------------------------------------------------------
typedef float        v8f   __attribute__((ext_vector_type(8)));
typedef __bf16       v16bf __attribute__((ext_vector_type(16)));
typedef unsigned int u32x4 __attribute__((ext_vector_type(4)));

union FragBF {
    u32x4 q[2];   // two 16-byte chunks (8 bf16 each)
    v16bf v;
};

// float -> bf16 bits, round-to-nearest-even
__device__ __forceinline__ unsigned short f2bf(float f) {
    unsigned int u = __float_as_uint(f);
    u += 0x7FFFu + ((u >> 16) & 1u);
    return (unsigned short)(u >> 16);
}

// ---------------------------------------------------------------------------
// Deterministic gaussian noise (hash + Box-Muller), keyed (stage,b,pixel).
// ---------------------------------------------------------------------------
__device__ __forceinline__ unsigned hash3(unsigned a, unsigned b, unsigned c) {
    unsigned h = a * 0x9E3779B9u ^ b * 0x85EBCA6Bu ^ c * 0xC2B2AE35u;
    h ^= h >> 16; h *= 0x7FEB352Du;
    h ^= h >> 15; h *= 0x846CA68Bu;
    h ^= h >> 16;
    return h;
}
__device__ __forceinline__ float gauss_noise(unsigned seed, int b, int p) {
    unsigned u1 = hash3(seed, (unsigned)b, (unsigned)p);
    unsigned u2 = hash3(seed ^ 0xDEADBEEFu, (unsigned)b, (unsigned)p);
    float f1 = ((float)u1 + 1.0f) * 2.3283064365386963e-10f;  // (0,1]
    float f2 = (float)u2 * 2.3283064365386963e-10f;           // [0,1)
    return sqrtf(-2.0f * logf(f1)) * cosf(6.283185307179586f * f2);
}

// ---------------------------------------------------------------------------
// style_std[b][c] = (style_base[b] * (1/16)) . affine_w[c] + affine_b[c] + 1
// ---------------------------------------------------------------------------
__global__ void style_affine_kernel(const float* __restrict__ sb,
                                    const float* __restrict__ aw,
                                    const float* __restrict__ ab,
                                    float* __restrict__ out, int Cin) {
    int i = blockIdx.x * blockDim.x + threadIdx.x;
    if (i >= 8 * Cin) return;
    int b = i / Cin, c = i % Cin;
    const float* s = sb + b * 512;
    const float* w = aw + (size_t)c * 512;
    float acc = 0.f;
    for (int k = 0; k < 512; ++k) acc += s[k] * w[k];
    out[i] = acc * 0.0625f + ab[c] + 1.0f;
}

// wsq[o][i] = sum_{k} w[o][i][k]^2
__global__ void wsq_kernel(const float* __restrict__ w, float* __restrict__ wsq,
                           int Cout, int Cin) {
    int i = blockIdx.x * blockDim.x + threadIdx.x;
    if (i >= Cout * Cin) return;
    const float* p = w + (size_t)i * 9;
    float s = 0.f;
#pragma unroll
    for (int j = 0; j < 9; ++j) s += p[j] * p[j];
    wsq[i] = s;
}

// demod[b][o] = rsqrt( sum_i wsq[o][i]*style[b][i]^2 + 1e-8 )
__global__ void demod_kernel(const float* __restrict__ wsq,
                             const float* __restrict__ style,
                             float* __restrict__ demod, int Cout, int Cin) {
    int i = blockIdx.x * blockDim.x + threadIdx.x;
    if (i >= 8 * Cout) return;
    int b = i / Cout, o = i % Cout;
    const float* wq = wsq + (size_t)o * Cin;
    const float* st = style + (size_t)b * Cin;
    float s = 1e-8f;
    for (int c = 0; c < Cin; ++c) { float t = st[c]; s += wq[c] * t * t; }
    demod[i] = rsqrtf(s);
}

// Weight prologue: convert to bf16 AND permute (O,I,3,3) -> [o][tap][ci].
// GEMM K index = tap*Cin + ci (each 32-chunk of K sits inside one tap).
__global__ void wcvt_kernel(const float* __restrict__ w,
                            unsigned short* __restrict__ wbf,
                            int Cout, int Cin) {
    int i = blockIdx.x * blockDim.x + threadIdx.x;
    if (i >= Cout * 9 * Cin) return;
    int ci   = i % Cin;
    int rest = i / Cin;
    int tap  = rest % 9;
    int o    = rest / 9;
    wbf[i] = f2bf(w[((size_t)(o * Cin + ci)) * 9 + tap]);
}

// Style-scale + layout change + HALO ZERO-PAD:
// NCHW f32 -> padded NHWC bf16 of shape (B, H+2, W+2, C), rim = 0.
// Interior: xs[((b*(H+2)+y+1)*(W+2)+x+1)*C + c] = x[b,c,y,x]*style[b][c]
__global__ void scale_x_pad_nhwc_kernel(const float* __restrict__ x,
                                        const float* __restrict__ style,
                                        unsigned short* __restrict__ xs,
                                        int B, int C, int H, int W, int bcast) {
    const int Hp = H + 2, Wp = W + 2;
    long long i = (long long)blockIdx.x * blockDim.x + threadIdx.x;
    if (i >= (long long)B * Hp * Wp * C) return;
    int c = (int)(i % C);
    long long rest = i / C;
    int pp = (int)(rest % (Hp * Wp));
    int b  = (int)(rest / (Hp * Wp));
    int yp = pp / Wp, xp = pp % Wp;
    int y = yp - 1, xx = xp - 1;
    unsigned short o = 0;
    if ((unsigned)y < (unsigned)H && (unsigned)xx < (unsigned)W) {
        int p = y * W + xx;
        float v = bcast ? x[(size_t)c * (H * W) + p]
                        : x[((size_t)b * C + c) * (H * W) + p];
        o = f2bf(v * style[b * C + c]);
    }
    xs[i] = o;
}

// ---------------------------------------------------------------------------
// Implicit-GEMM 3x3 SAME conv via v_wmma_f32_16x16x32_bf16.
// One wave32 computes a 16(M) x 32(N) output tile (2 accumulators, A reused).
// M = Cout, N = B*H*W, K = 9*Cin with K-order (tap, ci).
// Activations are halo-padded NHWC bf16, so ALL 9 taps are in-bounds:
// inner loop = 6 unconditional global_load_b128 + 2 v_wmma, no predication.
// ISA 16-bit fragment layout: lane<16: row=lane, K={0..7,16..23};
//                             lane>=16: row=lane-16, K={8..15,24..31}.
// Fused epilogue: demodulate, bias, noise, PReLU. Output NCHW f32.
// ---------------------------------------------------------------------------
__global__ __launch_bounds__(32)
void conv3x3_wmma_kernel(const unsigned short* __restrict__ xs,  // (B,H+2,W+2,Cin) bf16
                         const unsigned short* __restrict__ wbf, // (Cout, 9*Cin) bf16
                         const float* __restrict__ demod,
                         const float* __restrict__ bias,
                         const float* __restrict__ nscal,
                         const float* __restrict__ prelu_a,
                         float* __restrict__ out,                // (B,Cout,H,W) f32
                         int B, int Cin, int Cout, int H, int W, unsigned seed) {
    const int lane = threadIdx.x;
    const int row  = lane & 15;
    const int sel  = lane >> 4;
    const int oBase = blockIdx.x * 16;
    const int nBase = blockIdx.y * 32;
    const int HW = H * W;
    const int Hp = H + 2, Wp = W + 2;

    // A row for this lane (M = oBase+row)
    const unsigned short* arow = wbf + (size_t)(oBase + row) * 9 * Cin;

    // Two pixel groups for this lane (N = nBase+row, nBase+16+row)
    const int n0 = nBase + row;
    const int n1 = nBase + 16 + row;
    const int b0 = n0 / HW, p0 = n0 % HW, y0 = p0 / W, x0 = p0 % W;
    const int b1 = n1 / HW, p1 = n1 % HW, y1 = p1 / W, x1 = p1 % W;

    const int koff = sel * 8;  // lane's K sub-chunk offset

    v8f acc0 = {};
    v8f acc1 = {};

#pragma unroll
    for (int tap = 0; tap < 9; ++tap) {
        const int dy = tap / 3;       // 0..2 (halo origin shift: y0+dy indexes padded)
        const int dx = tap % 3;       // 0..2
        const unsigned short* bp0 =
            xs + ((size_t)(b0 * Hp + y0 + dy) * Wp + (x0 + dx)) * Cin + koff;
        const unsigned short* bp1 =
            xs + ((size_t)(b1 * Hp + y1 + dy) * Wp + (x1 + dx)) * Cin + koff;
        const unsigned short* ap = arow + tap * Cin + koff;

        for (int c0 = 0; c0 < Cin; c0 += 32) {
            FragBF fa, fb0, fb1;
            fa.q[0]  = *(const u32x4*)(ap + c0);
            fa.q[1]  = *(const u32x4*)(ap + c0 + 16);
            fb0.q[0] = *(const u32x4*)(bp0 + c0);
            fb0.q[1] = *(const u32x4*)(bp0 + c0 + 16);
            fb1.q[0] = *(const u32x4*)(bp1 + c0);
            fb1.q[1] = *(const u32x4*)(bp1 + c0 + 16);
            acc0 = __builtin_amdgcn_wmma_f32_16x16x32_bf16(
                false, fa.v, false, fb0.v, (short)0, acc0, false, false);
            acc1 = __builtin_amdgcn_wmma_f32_16x16x32_bf16(
                false, fa.v, false, fb1.v, (short)0, acc1, false, false);
        }
    }

    // Epilogue. C/D layout: lane<16 -> N=lane, M=r; lane>=16 -> M=r+8.
    const float a = prelu_a[0];
    {
        const float nz = gauss_noise(seed, b0, p0);
#pragma unroll
        for (int r = 0; r < 8; ++r) {
            int o = oBase + r + sel * 8;
            float v = acc0[r] * demod[b0 * Cout + o] + bias[o] + nscal[o] * nz;
            v = (v >= 0.f) ? v : a * v;
            out[((size_t)(b0 * Cout + o) * H + y0) * W + x0] = v;
        }
    }
    {
        const float nz = gauss_noise(seed, b1, p1);
#pragma unroll
        for (int r = 0; r < 8; ++r) {
            int o = oBase + r + sel * 8;
            float v = acc1[r] * demod[b1 * Cout + o] + bias[o] + nscal[o] * nz;
            v = (v >= 0.f) ? v : a * v;
            out[((size_t)(b1 * Cout + o) * H + y1) * W + x1] = v;
        }
    }
}

// ---------------------------------------------------------------------------
// Bilinear 2x upsample, both align_corners modes (NCHW f32).
// ---------------------------------------------------------------------------
__global__ void upsample2x_kernel(const float* __restrict__ in, float* __restrict__ out,
                                  int B, int C, int H, int W, int align) {
    int Ho = 2 * H, Wo = 2 * W;
    long long i = (long long)blockIdx.x * blockDim.x + threadIdx.x;
    long long total = (long long)B * C * Ho * Wo;
    if (i >= total) return;
    int xo = (int)(i % Wo);
    int yo = (int)((i / Wo) % Ho);
    long long bc = i / ((long long)Wo * Ho);

    float py, px;
    if (align) {
        py = (float)yo * (float)(H - 1) / (float)(Ho - 1);
        px = (float)xo * (float)(W - 1) / (float)(Wo - 1);
    } else {
        py = fminf(fmaxf(((float)yo + 0.5f) * 0.5f - 0.5f, 0.f), (float)(H - 1));
        px = fminf(fmaxf(((float)xo + 0.5f) * 0.5f - 0.5f, 0.f), (float)(W - 1));
    }
    int y0 = (int)floorf(py), x0 = (int)floorf(px);
    int y1 = min(y0 + 1, H - 1), x1 = min(x0 + 1, W - 1);
    float wy = py - (float)y0, wx = px - (float)x0;

    const float* src = in + bc * (long long)H * W;
    float v00 = src[y0 * W + x0], v01 = src[y0 * W + x1];
    float v10 = src[y1 * W + x0], v11 = src[y1 * W + x1];
    float v0 = v00 * (1.f - wx) + v01 * wx;
    float v1 = v10 * (1.f - wx) + v11 * wx;
    out[i] = v0 * (1.f - wy) + v1 * wy;
}

// skip: t[b,o,p] = (sum_i xup[b,i,p]*cw[o,i] + cb[o]) / Ci
__global__ void conv1x1_skip_kernel(const float* __restrict__ xup,
                                    const float* __restrict__ cw,
                                    const float* __restrict__ cb,
                                    float* __restrict__ t,
                                    int B, int Ci, int Co, int HW) {
    long long i = (long long)blockIdx.x * blockDim.x + threadIdx.x;
    if (i >= (long long)B * Co * HW) return;
    int p = (int)(i % HW);
    int o = (int)((i / HW) % Co);
    int b = (int)(i / ((long long)HW * Co));
    const float* xr = xup + (long long)b * Ci * HW + p;
    const float* wr = cw + (size_t)o * Ci;
    float s = 0.f;
    for (int c = 0; c < Ci; ++c) s += xr[(long long)c * HW] * wr[c];
    t[i] = (s + cb[o]) / (float)Ci;
}

// x = (x + t) / ROOT_2  (reference uses literal 1.41421)
__global__ void add_skip_kernel(float* __restrict__ x, const float* __restrict__ t,
                                long long n) {
    long long i = (long long)blockIdx.x * blockDim.x + threadIdx.x;
    if (i >= n) return;
    x[i] = (x[i] + t[i]) * (1.0f / 1.41421f);
}

// rgb = clip( (x*sqrt(2/C)) . out_w + out_b , 0, 1);  img (set or add)
__global__ void to_rgb_kernel(const float* __restrict__ x,
                              const float* __restrict__ ow,
                              const float* __restrict__ ob,
                              float* __restrict__ img,
                              int B, int C, int HW, int accum) {
    long long i = (long long)blockIdx.x * blockDim.x + threadIdx.x;
    if (i >= (long long)B * 3 * HW) return;
    int p = (int)(i % HW);
    int o = (int)((i / HW) % 3);
    int b = (int)(i / ((long long)HW * 3));
    float sc = sqrtf(2.0f / (float)C);
    const float* xr = x + (long long)b * C * HW + p;
    const float* wr = ow + (size_t)o * C;
    float s = 0.f;
    for (int c = 0; c < C; ++c) s += xr[(long long)c * HW] * wr[c];
    float v = s * sc + ob[o];
    v = fminf(fmaxf(v, 0.f), 1.f);
    img[i] = accum ? (img[i] + v) : v;
}

__global__ void final_clip_kernel(const float* __restrict__ img,
                                  float* __restrict__ out, long long n) {
    long long i = (long long)blockIdx.x * blockDim.x + threadIdx.x;
    if (i >= n) return;
    out[i] = fminf(fmaxf(img[i], 0.f), 1.f);
}

// ---------------------------------------------------------------------------
// Host orchestration
// ---------------------------------------------------------------------------
static inline int cdiv(long long a, int b) { return (int)((a + b - 1) / b); }

struct BlkPtrs {
    const float *aw, *ab, *cw, *cb, *ns, *pa, *ow, *ob;
};

extern "C" void kernel_launch(void* const* d_in, const int* in_sizes, int n_in,
                              void* d_out, int out_size, void* d_ws, size_t ws_size,
                              hipStream_t stream) {
    (void)in_sizes; (void)n_in; (void)out_size; (void)ws_size;
    // ---- input unpack (dict flattened depth-first) ----
    int idx = 0;
    const float* style_base    = (const float*)d_in[idx++];   // (8,512)
    const float* basic_texture = (const float*)d_in[idx++];   // (512,4,4)
    BlkPtrs blk[13];
    for (int b = 0; b < 13; ++b) {
        bool has_out = (b == 0) || (((b - 1) & 1) == 1);  // block0 + every LATTER
        blk[b].aw = (const float*)d_in[idx++];
        blk[b].ab = (const float*)d_in[idx++];
        blk[b].cw = (const float*)d_in[idx++];
        blk[b].cb = (const float*)d_in[idx++];
        blk[b].ns = (const float*)d_in[idx++];
        blk[b].pa = (const float*)d_in[idx++];
        if (has_out) {
            blk[b].ow = (const float*)d_in[idx++];
            blk[b].ob = (const float*)d_in[idx++];
        } else { blk[b].ow = nullptr; blk[b].ob = nullptr; }
    }
    const float* c1w[6]; const float* c1b[6];
    for (int i = 0; i < 6; ++i) {
        c1w[i] = (const float*)d_in[idx++];
        c1b[i] = (const float*)d_in[idx++];
    }

    // ---- workspace carve (in floats) ----
    const size_t NMAX = 16777216;  // 8*32*256*256 (largest activation, floats)
    float* ws   = (float*)d_ws;
    float* X0   = ws;                        // current feature x / latter out (NCHW f32)
    float* X1   = ws + NMAX;                 // upsampled x (NCHW f32)
    float* X2   = ws + 2 * NMAX;             // styled padded NHWC bf16 scratch
    float* X3   = ws + 3 * NMAX;             // former conv out (NCHW f32)
    float* TB   = ws + 4 * NMAX;             // skip t (<= NMAX/2)
    float* IMGA = ws + 4 * NMAX + NMAX / 2;  // rgb accumulators
    float* IMGB = IMGA + 2097152;
    float* STYL = IMGB + 2097152;            // 8*512
    float* DEMO = STYL + 4096;               // 8*512
    float* WSQ  = DEMO + 4096;               // 512*512
    unsigned short* WBF = (unsigned short*)(WSQ + 262144);  // 512*9*512 bf16
    unsigned short* XS  = (unsigned short*)X2;              // padded styled input
    // XS worst case: 8*258*258*32 bf16 = 34 MB < 64 MiB slot; single buffer is
    // enough because each conv fully consumes it before the next scale kernel.

    const int B = 8;
    const int EL = 256;

    // style -> padded NHWC bf16 scale -> weight cvt/permute -> wsq/demod -> conv
    auto run_modconv = [&](const float* xin, int bcast, const BlkPtrs& p,
                           int Cin, int Cout, int H, int W,
                           float* yout, unsigned seed) {
        int HW = H * W;
        long long npad = (long long)B * (H + 2) * (W + 2) * Cin;
        style_affine_kernel<<<cdiv((long long)B * Cin, EL), EL, 0, stream>>>(
            style_base, p.aw, p.ab, STYL, Cin);
        scale_x_pad_nhwc_kernel<<<cdiv(npad, EL), EL, 0, stream>>>(
            xin, STYL, XS, B, Cin, H, W, bcast);
        wcvt_kernel<<<cdiv((long long)Cout * 9 * Cin, EL), EL, 0, stream>>>(
            p.cw, WBF, Cout, Cin);
        wsq_kernel<<<cdiv((long long)Cout * Cin, EL), EL, 0, stream>>>(
            p.cw, WSQ, Cout, Cin);
        demod_kernel<<<cdiv((long long)B * Cout, EL), EL, 0, stream>>>(
            WSQ, STYL, DEMO, Cout, Cin);
        dim3 grid(Cout / 16, (B * HW) / 32);
        conv3x3_wmma_kernel<<<grid, 32, 0, stream>>>(
            XS, WBF, DEMO, p.cb, p.ns, p.pa, yout, B, Cin, Cout, H, W, seed);
    };

    // ---- block 0: 4x4, 512->512, out=True ----
    run_modconv(basic_texture, /*bcast=*/1, blk[0], 512, 512, 4, 4, X0, 0u);
    to_rgb_kernel<<<cdiv((long long)B * 3 * 16, EL), EL, 0, stream>>>(
        X0, blk[0].ow, blk[0].ob, IMGA, B, 512, 16, /*accum=*/0);
    upsample2x_kernel<<<cdiv((long long)B * 3 * 64, EL), EL, 0, stream>>>(
        IMGA, IMGB, B, 3, 4, 4, /*align=*/1);
    float* imgCur = IMGB;
    float* imgAlt = IMGA;

    const int ins[6]  = {512, 512, 256, 128, 64, 32};
    const int outs[6] = {512, 256, 128,  64, 32, 16};

    for (int i = 0; i < 6; ++i) {
        int ci = ins[i], co = outs[i];
        int hin = 4 << i, hout = 8 << i;
        int HWo = hout * hout;
        const BlkPtrs& former = blk[1 + 2 * i];
        const BlkPtrs& latter = blk[2 + 2 * i];

        // shared align_corners=False 2x upsample (used by both skip and former)
        upsample2x_kernel<<<cdiv((long long)B * ci * HWo, EL), EL, 0, stream>>>(
            X0, X1, B, ci, hin, hin, /*align=*/0);
        // skip: t = conv1x1(xup)/ci
        conv1x1_skip_kernel<<<cdiv((long long)B * co * HWo, EL), EL, 0, stream>>>(
            X1, c1w[i], c1b[i], TB, B, ci, co, HWo);
        // former: modconv ci->ci on xup
        run_modconv(X1, 0, former, ci, ci, hout, hout, X3,
                    (unsigned)(2 * i + 1));
        // latter: modconv ci->co on former output (X0 free -> reuse as out)
        run_modconv(X3, 0, latter, ci, co, hout, hout, X0,
                    (unsigned)(2 * i + 2));
        // x = (x + t)/sqrt(2)
        add_skip_kernel<<<cdiv((long long)B * co * HWo, EL), EL, 0, stream>>>(
            X0, TB, (long long)B * co * HWo);
        // img += rgb(x)
        to_rgb_kernel<<<cdiv((long long)B * 3 * HWo, EL), EL, 0, stream>>>(
            X0, latter.ow, latter.ob, imgCur, B, co, HWo, /*accum=*/1);
        if (i < 5) {
            upsample2x_kernel<<<cdiv((long long)B * 3 * 4 * HWo, EL), EL, 0, stream>>>(
                imgCur, imgAlt, B, 3, hout, hout, /*align=*/1);
            float* tmp = imgCur; imgCur = imgAlt; imgAlt = tmp;
        }
    }

    // final clip -> d_out (8,3,256,256)
    long long nOut = (long long)B * 3 * 256 * 256;
    final_clip_kernel<<<cdiv(nOut, EL), EL, 0, stream>>>(imgCur, (float*)d_out, nOut);
}
